// MLPPredictor_15960098471964
// MI455X (gfx1250) — matrix operand order repaired and checked by reading, not compiled, not run on hardware
//
#include <hip/hip_runtime.h>
#include <hip/hip_bf16.h>
#include <cstdint>

typedef __attribute__((ext_vector_type(16))) __bf16 v16bf;
typedef __attribute__((ext_vector_type(8)))  __bf16 v8bf;
typedef __attribute__((ext_vector_type(8)))  float  v8f;

#define IN_FEAT 128
#define HID     128
#define NT      8      // N tiles of 16 (128/16)
#define KC1     8      // K chunks of 32 for layer 1 (256/32)
#define KC2     4      // K chunks of 32 for layers 2/3 (128/32)
#define WAVES   6
#define TPB     (WAVES * 32)
#define MT      2      // M tiles (16 edges each) sharing every B fragment

__device__ __forceinline__ void pack8(v16bf& a, int base, float4 lo, float4 hi) {
  a[base + 0] = (__bf16)lo.x; a[base + 1] = (__bf16)lo.y;
  a[base + 2] = (__bf16)lo.z; a[base + 3] = (__bf16)lo.w;
  a[base + 4] = (__bf16)hi.x; a[base + 5] = (__bf16)hi.y;
  a[base + 6] = (__bf16)hi.z; a[base + 7] = (__bf16)hi.w;
}

// LDS (149.5KB) pins us at 2 blocks/WGP = 3 waves/SIMD, so let the register
// allocator use the full 1024/3 VGPR budget for deeper load pipelining.
__global__ __launch_bounds__(TPB, 3) void edge_mlp_wmma_kernel(
    const float* __restrict__ h,
    const long long* __restrict__ src,
    const long long* __restrict__ dst,
    const float* __restrict__ W1, const float* __restrict__ b1,
    const float* __restrict__ W2, const float* __restrict__ b2,
    const float* __restrict__ W3, const float* __restrict__ b3,
    float* __restrict__ out, int E)
{
  // Weights staged in bf16, pre-swizzled into WMMA B-fragment order:
  // slot = ((ntile*KC + kchunk)*32 + lane)*16 + j,
  // lane = (((k>>3)&1)<<4)|col, j = (k&7)|((k>>4)<<3)   (B 32x16: lanes 0-15 K=0..7,16..23; 16-31 K=8..15,24..31)
  __shared__ __align__(32) __bf16 W1s[NT * KC1 * 32 * 16];   // 64 KB
  __shared__ __align__(32) __bf16 W2s[NT * KC2 * 32 * 16];   // 32 KB
  __shared__ __align__(32) __bf16 W3s[KC2 * 32 * 16];        // 4 KB, N padded 2->16 (zeros)
  __shared__ float b1s[HID], b2s[HID], b3s[2];
  __shared__ __align__(32) __bf16 xbuf[WAVES][MT][16 * HID]; // 48 KB, per-wave 2x(16x128)

  const int tid = threadIdx.x;

  // ---- one-time stage: convert weights f32 -> bf16 into fragment-ready LDS
  for (int i = tid; i < 2 * IN_FEAT * HID; i += TPB) {
    int k = i >> 7, n = i & 127;
    int ntile = n >> 4, col = n & 15;
    int kchunk = k >> 5, kk = k & 31;
    int fl = (((kk >> 3) & 1) << 4) | col;
    W1s[(((ntile * KC1 + kchunk) * 32) + fl) * 16 + ((kk & 7) | ((kk >> 4) << 3))] = (__bf16)W1[i];
  }
  for (int i = tid; i < HID * HID; i += TPB) {
    int k = i >> 7, n = i & 127;
    int ntile = n >> 4, col = n & 15;
    int kchunk = k >> 5, kk = k & 31;
    int fl = (((kk >> 3) & 1) << 4) | col;
    W2s[(((ntile * KC2 + kchunk) * 32) + fl) * 16 + ((kk & 7) | ((kk >> 4) << 3))] = (__bf16)W2[i];
  }
  // W3: 128x2 padded to 128x16 B-fragments (cols 2..15 = 0)
  for (int i = tid; i < KC2 * 32 * 16; i += TPB) {
    int kchunk = i >> 9;
    int fl = (i >> 4) & 31;
    int j = i & 15;
    int col = fl & 15;
    int k = kchunk * 32 + ((j >> 3) << 4) + ((fl >> 4) << 3) + (j & 7);
    W3s[i] = (col < 2) ? (__bf16)W3[k * 2 + col] : (__bf16)0.0f;
  }
  if (tid < HID) { b1s[tid] = b1[tid]; b2s[tid] = b2[tid]; }
  if (tid < 2)   { b3s[tid] = b3[tid]; }
  __syncthreads();

  const int lane = tid & 31;
  const int wave = tid >> 5;
  const int row  = lane & 15;          // A-fragment row this lane holds
  const int ncol = lane & 15;          // B/D column this lane holds
  const int hi   = lane >> 4;          // lane half
  const int off  = hi << 3;            // A K-octet offset within chunk: 0 or 8

  const int numTiles = (E + 15) >> 4;
  const int numPairs = (numTiles + MT - 1) / MT;
  const int gwave  = blockIdx.x * WAVES + wave;
  const int nwaves = gridDim.x * WAVES;

  for (int pair = gwave; pair < numPairs; pair += nwaves) {
    const int ebase0 = (pair * MT) << 4;
    const float* hu[MT];
    const float* hv[MT];
    #pragma unroll
    for (int t = 0; t < MT; ++t) {
      int e = ebase0 + (t << 4) + row; if (e >= E) e = E - 1;  // clamp, keep EXEC full
      hu[t] = h + (size_t)src[e] * IN_FEAT;
      hv[t] = h + (size_t)dst[e] * IN_FEAT;
    }

    // ---- layer 1: kc-outer; one B fragment feeds both M tiles (B traffic halved)
    v8f acc1[MT][NT];
    #pragma unroll
    for (int t = 0; t < MT; ++t)
      #pragma unroll
      for (int nt = 0; nt < NT; ++nt) acc1[t][nt] = (v8f){};  // inline-0 C at kc==0

    #pragma unroll
    for (int kc = 0; kc < KC1; ++kc) {
      int k0 = kc * 32 + off;
      int k1 = k0 + 16;
      v16bf a[MT];
      #pragma unroll
      for (int t = 0; t < MT; ++t) {
        const float* p0 = (k0 < IN_FEAT) ? (hu[t] + k0) : (hv[t] + (k0 - IN_FEAT));
        const float* p1 = (k1 < IN_FEAT) ? (hu[t] + k1) : (hv[t] + (k1 - IN_FEAT));
        float4 f0 = *(const float4*)p0;
        float4 f1 = *(const float4*)(p0 + 4);
        float4 f2 = *(const float4*)p1;
        float4 f3 = *(const float4*)(p1 + 4);
        pack8(a[t], 0, f0, f1);
        pack8(a[t], 8, f2, f3);
      }
      // software one-fragment-ahead B prefetch: load B[nt+1] before using B[nt]
      v16bf bn = *(const v16bf*)&W1s[((0 * KC1 + kc) * 32 + lane) * 16];
      #pragma unroll
      for (int nt = 0; nt < NT; ++nt) {
        v16bf b = bn;
        if (nt + 1 < NT)
          bn = *(const v16bf*)&W1s[(((nt + 1) * KC1 + kc) * 32 + lane) * 16];
        #pragma unroll
        for (int t = 0; t < MT; ++t)
          acc1[t][nt] = __builtin_amdgcn_wmma_f32_16x16x32_bf16(
                            false, a[t], false, b, (short)0, acc1[t][nt], false, false);
      }
    }
    #pragma unroll
    for (int t = 0; t < MT; ++t) {
      __bf16* xw = &xbuf[wave][t][0];
      #pragma unroll
      for (int nt = 0; nt < NT; ++nt) {
        int n = (nt << 4) | ncol;
        float bias = b1s[n];
        #pragma unroll
        for (int r = 0; r < 8; ++r) {            // D: VGPR r -> row r + 8*half
          float v = acc1[t][nt][r] + bias;
          v = v > 0.f ? v : 0.f;
          xw[(r + (hi << 3)) * HID + n] = (__bf16)v;
        }
      }
    }

    // ---- layer 2: A from LDS (same-wave DS ops are in-order); shared B
    v8f acc2[MT][NT];
    #pragma unroll
    for (int t = 0; t < MT; ++t)
      #pragma unroll
      for (int nt = 0; nt < NT; ++nt) acc2[t][nt] = (v8f){};
    #pragma unroll
    for (int kc = 0; kc < KC2; ++kc) {
      int k0 = kc * 32 + off;
      v16bf a[MT];
      #pragma unroll
      for (int t = 0; t < MT; ++t) {
        const __bf16* xw = &xbuf[wave][t][0];
        v8bf lo = *(const v8bf*)&xw[row * HID + k0];
        v8bf hx = *(const v8bf*)&xw[row * HID + k0 + 16];
        a[t] = __builtin_shufflevector(lo, hx, 0,1,2,3,4,5,6,7,8,9,10,11,12,13,14,15);
      }
      v16bf bn = *(const v16bf*)&W2s[((0 * KC2 + kc) * 32 + lane) * 16];
      #pragma unroll
      for (int nt = 0; nt < NT; ++nt) {
        v16bf b = bn;
        if (nt + 1 < NT)
          bn = *(const v16bf*)&W2s[(((nt + 1) * KC2 + kc) * 32 + lane) * 16];
        #pragma unroll
        for (int t = 0; t < MT; ++t)
          acc2[t][nt] = __builtin_amdgcn_wmma_f32_16x16x32_bf16(
                            false, a[t], false, b, (short)0, acc2[t][nt], false, false);
      }
    }
    #pragma unroll
    for (int t = 0; t < MT; ++t) {
      __bf16* xw = &xbuf[wave][t][0];
      #pragma unroll
      for (int nt = 0; nt < NT; ++nt) {
        int n = (nt << 4) | ncol;
        float bias = b2s[n];
        #pragma unroll
        for (int r = 0; r < 8; ++r) {
          float v = acc2[t][nt][r] + bias;
          v = v > 0.f ? v : 0.f;
          xw[(r + (hi << 3)) * HID + n] = (__bf16)v;
        }
      }
    }

    // ---- layer 3 (OUT=2 padded to one 16-wide N tile): 4 WMMAs per M tile, shared B
    {
      v8f acc[MT];
      #pragma unroll
      for (int t = 0; t < MT; ++t) acc[t] = (v8f){};
      v16bf bn = *(const v16bf*)&W3s[(0 * 32 + lane) * 16];
      #pragma unroll
      for (int kc = 0; kc < KC2; ++kc) {
        int k0 = kc * 32 + off;
        v16bf b = bn;
        if (kc + 1 < KC2)
          bn = *(const v16bf*)&W3s[((kc + 1) * 32 + lane) * 16];
        #pragma unroll
        for (int t = 0; t < MT; ++t) {
          const __bf16* xw = &xbuf[wave][t][0];
          v8bf lo = *(const v8bf*)&xw[row * HID + k0];
          v8bf hx = *(const v8bf*)&xw[row * HID + k0 + 16];
          v16bf a = __builtin_shufflevector(lo, hx, 0,1,2,3,4,5,6,7,8,9,10,11,12,13,14,15);
          acc[t] = __builtin_amdgcn_wmma_f32_16x16x32_bf16(
                       false, a, false, b, (short)0, acc[t], false, false);
        }
      }
      // only cols 0,1 are real outputs: lanes 0,1 (rows 0-7) and 16,17 (rows 8-15)
      if (ncol < 2) {
        float bias = b3s[ncol];
        #pragma unroll
        for (int t = 0; t < MT; ++t) {
          int ebase = ebase0 + (t << 4);
          #pragma unroll
          for (int r = 0; r < 8; ++r) {
            int eo = ebase + r + (hi << 3);
            if (eo < E) out[eo * 2 + ncol] = acc[t][r] + bias;
          }
        }
      }
    }
  }
}

extern "C" void kernel_launch(void* const* d_in, const int* in_sizes, int n_in,
                              void* d_out, int out_size, void* d_ws, size_t ws_size,
                              hipStream_t stream) {
  const float*     h   = (const float*)d_in[0];
  const long long* src = (const long long*)d_in[1];   // int64 per reference
  const long long* dst = (const long long*)d_in[2];
  const float*     W1  = (const float*)d_in[3];
  const float*     b1  = (const float*)d_in[4];
  const float*     W2  = (const float*)d_in[5];
  const float*     b2  = (const float*)d_in[6];
  const float*     W3  = (const float*)d_in[7];
  const float*     b3  = (const float*)d_in[8];
  float* out = (float*)d_out;
  int E = in_sizes[1];
  if (E <= 0) return;

  int numTiles = (E + 15) >> 4;
  int numPairs = (numTiles + MT - 1) / MT;
  int blocks = (numPairs + WAVES - 1) / WAVES;
  if (blocks > 512) blocks = 512;    // 2 blocks/WGP at 149.5KB LDS; grid-stride covers rest
  edge_mlp_wmma_kernel<<<blocks, TPB, 0, stream>>>(h, src, dst, W1, b1, W2, b2, W3, b3, out, E);
}
